// GNN_64355789963872
// MI455X (gfx1250) — matrix operand order repaired and checked by reading, not compiled, not run on hardware
//
#include <hip/hip_runtime.h>
#include <hip/hip_bf16.h>

#define NN 50000
#define HC 128
#define NH 4
#define NG 64

typedef __attribute__((ext_vector_type(16))) _Float16 v16h;
typedef __attribute__((ext_vector_type(8)))  float    v8f;

// ---------------- ordered-float <-> uint key (for atomic max) ----------------
__device__ __forceinline__ unsigned fkey(float f) {
    unsigned u = __float_as_uint(f);
    return (u & 0x80000000u) ? ~u : (u | 0x80000000u);
}
__device__ __forceinline__ float funkey(unsigned k) {
    unsigned u = (k & 0x80000000u) ? (k ^ 0x80000000u) : ~k;
    return __uint_as_float(u);
}

__device__ __forceinline__ float4 relu4(float4 v) {
    v.x = fmaxf(v.x, 0.f); v.y = fmaxf(v.y, 0.f);
    v.z = fmaxf(v.z, 0.f); v.w = fmaxf(v.w, 0.f);
    return v;
}

// ---------------- pack W[K,128] into per-lane WMMA B-fragment layout (f16) ----
// wf[((chunk*8 + tile_n)*32 + lane)*16 + e] = W[(chunk*32 + (e&8)*2 + (lane>>4)*8 + (e&7))*128
//                                               + tile_n*16 + (lane&15)]
__global__ void pack_wfrag(const float* __restrict__ W, _Float16* __restrict__ wf, int K) {
    int i = blockIdx.x * blockDim.x + threadIdx.x;
    if (i >= K * HC) return;
    int e     = i & 15;
    int lane  = (i >> 4) & 31;
    int tn    = (i >> 9) & 7;
    int chunk = i >> 12;
    int k   = chunk * 32 + ((e & 8) << 1) + ((lane >> 4) << 3) + (e & 7);
    int col = tn * 16 + (lane & 15);
    wf[i] = (_Float16)W[k * HC + col];
}

// ---------------- WMMA GEMM: out[M,128] = (RELU? relu(A) : A)[M,K] @ W[K,128]
// one wave computes one 16x16 tile; A via 4x b128 loads, B via one v16h load.
template <int K, int RELU>
__global__ void gemm_wmma(const float* __restrict__ A, const _Float16* __restrict__ wf,
                          float* __restrict__ out, int M) {
    int gwave = (blockIdx.x * blockDim.x + threadIdx.x) >> 5;
    int lane  = threadIdx.x & 31;
    int tile_m = gwave >> 3;          // 8 N-tiles (128/16)
    int tile_n = gwave & 7;
    if (tile_m * 16 >= M) return;     // wave-uniform: EXEC stays all-ones

    int l15 = lane & 15;
    int hi  = lane >> 4;              // 0 or 1
    int row = tile_m * 16 + l15;      // A row for this lane
    int col = tile_n * 16 + l15;      // D column for this lane

    v8f acc = {};
    #pragma unroll
    for (int k0 = 0; k0 < K; k0 += 32) {
        // ---- A fragment: two contiguous 8-float runs -> 4x float4 loads ----
        const float* Ar = A + (long long)row * K + k0 + hi * 8;
        float4 a0 = *(const float4*)(Ar);
        float4 a1 = *(const float4*)(Ar + 4);
        float4 a2 = *(const float4*)(Ar + 16);
        float4 a3 = *(const float4*)(Ar + 20);
        if (RELU) { a0 = relu4(a0); a1 = relu4(a1); a2 = relu4(a2); a3 = relu4(a3); }
        v16h a;
        a[0]=(_Float16)a0.x; a[1]=(_Float16)a0.y; a[2]=(_Float16)a0.z; a[3]=(_Float16)a0.w;
        a[4]=(_Float16)a1.x; a[5]=(_Float16)a1.y; a[6]=(_Float16)a1.z; a[7]=(_Float16)a1.w;
        a[8]=(_Float16)a2.x; a[9]=(_Float16)a2.y; a[10]=(_Float16)a2.z; a[11]=(_Float16)a2.w;
        a[12]=(_Float16)a3.x; a[13]=(_Float16)a3.y; a[14]=(_Float16)a3.z; a[15]=(_Float16)a3.w;
        // ---- B fragment: pre-packed, one 32B coalesced load ----
        v16h b = *(const v16h*)(wf + ((((k0 >> 5) * 8 + tile_n) * 32 + lane) << 4));
        acc = __builtin_amdgcn_wmma_f32_16x16x32_f16(false, a, false, b,
                                                     (short)0, acc, false, false);
    }
    // D: VGPR r -> M = r + hi*8
    #pragma unroll
    for (int r = 0; r < 8; ++r)
        out[(long long)(tile_m * 16 + r + hi * 8) * HC + col] = acc[r];
}

// ---------------- per-(node,head) attention coefficients ----------------
__global__ void attn_coeff(const float* __restrict__ xs, const float* __restrict__ as_,
                           const float* __restrict__ ad_, float* __restrict__ a_src,
                           float* __restrict__ a_dst, int N) {
    int i = blockIdx.x * blockDim.x + threadIdx.x;
    if (i >= N * NH) return;
    int n = i >> 2, h = i & 3;
    const float4* xp = (const float4*)(xs + (long long)n * HC + h * 32);
    const float4* sp = (const float4*)(as_ + h * 32);
    const float4* dp = (const float4*)(ad_ + h * 32);
    float s = 0.f, d = 0.f;
    #pragma unroll
    for (int q = 0; q < 8; ++q) {
        float4 v = xp[q], a = sp[q], b = dp[q];
        s += v.x*a.x + v.y*a.y + v.z*a.z + v.w*a.w;
        d += v.x*b.x + v.y*b.y + v.z*b.z + v.w*b.w;
    }
    a_src[i] = s; a_dst[i] = d;
}

// ---------------- edge-attr self-loop mean (fill_value='mean') ----------------
__global__ void edge_attr_stats(const int* __restrict__ dstv, const float* __restrict__ attr,
                                float* __restrict__ lsum, float* __restrict__ lcnt, int E) {
    int e = blockIdx.x * blockDim.x + threadIdx.x;
    if (e >= E) return;
    int d = dstv[e];
    atomicAdd(&lsum[d], attr[e]);
    atomicAdd(&lcnt[d], 1.f);
}
__global__ void loop_attr_final(const float* __restrict__ lsum, const float* __restrict__ lcnt,
                                float* __restrict__ lattr, int N) {
    int n = blockIdx.x * blockDim.x + threadIdx.x;
    if (n < N) lattr[n] = lsum[n] / fmaxf(lcnt[n], 1.f);
}
// coef[h] = sum_c We[0, h*32+c] * ae[h,c]   (EDGE_DIM==1 collapses edge term)
__global__ void compute_ecoef(const float* __restrict__ We, const float* __restrict__ ae,
                              float* __restrict__ coef) {
    int h = threadIdx.x;
    if (h >= NH) return;
    float s = 0.f;
    #pragma unroll
    for (int c = 0; c < 32; ++c) s += We[h*32+c] * ae[h*32+c];
    coef[h] = s;
}

// ---------------- raw (pre-softmax) alpha for edge e, head h ----------------
__device__ __forceinline__ float raw_alpha(long long e, int h, const int* srcv, const int* dstv,
                                           const float* attr, const float* lattr,
                                           const float* a_src, const float* a_dst,
                                           const float* ecoef, long long E, int& s, int& d) {
    float ea = 0.f;
    if (e < E) { s = srcv[e]; d = dstv[e]; if (attr) ea = attr[e]; }
    else       { s = d = (int)(e - E);     if (attr) ea = lattr[d]; }
    float al = a_src[s*NH + h] + a_dst[d*NH + h];
    if (attr) al += ea * ecoef[h];
    return al > 0.f ? al : 0.2f * al;   // leaky relu
}

__global__ void alpha_max(const int* srcv, const int* dstv, const float* attr, const float* lattr,
                          const float* a_src, const float* a_dst, const float* ecoef,
                          unsigned* __restrict__ amax, long long E, int N) {
    long long i = (long long)blockIdx.x * blockDim.x + threadIdx.x;
    if (i >= (E + N) * NH) return;
    int h = (int)(i & 3); long long e = i >> 2;
    int s, d;
    float al = raw_alpha(e, h, srcv, dstv, attr, lattr, a_src, a_dst, ecoef, E, s, d);
    atomicMax(&amax[d*NH + h], fkey(al));
}

__global__ void alpha_denom(const int* srcv, const int* dstv, const float* attr, const float* lattr,
                            const float* a_src, const float* a_dst, const float* ecoef,
                            const unsigned* __restrict__ amax, float* __restrict__ denom,
                            long long E, int N) {
    long long i = (long long)blockIdx.x * blockDim.x + threadIdx.x;
    if (i >= (E + N) * NH) return;
    int h = (int)(i & 3); long long e = i >> 2;
    int s, d;
    float al = raw_alpha(e, h, srcv, dstv, attr, lattr, a_src, a_dst, ecoef, E, s, d);
    atomicAdd(&denom[d*NH + h], __expf(al - funkey(amax[d*NH + h])));
}

// ---------------- normalized message scatter: 128 threads per edge ----------------
// each 32-lane wave maps to one (edge, head): scalar reads broadcast.
__global__ void message_pass(const int* srcv, const int* dstv, const float* attr, const float* lattr,
                             const float* a_src, const float* a_dst, const float* ecoef,
                             const unsigned* __restrict__ amax, const float* __restrict__ denom,
                             const float* __restrict__ xs, float* __restrict__ out,
                             long long E, int N) {
    long long i = (long long)blockIdx.x * blockDim.x + threadIdx.x;
    if (i >= (E + N) * HC) return;
    int c = (int)(i & 127); long long e = i >> 7; int h = c >> 5;
    int s, d;
    float al = raw_alpha(e, h, srcv, dstv, attr, lattr, a_src, a_dst, ecoef, E, s, d);
    float alpha = __expf(al - funkey(amax[d*NH + h])) / (denom[d*NH + h] + 1e-16f);
    atomicAdd(&out[(long long)d * HC + c], xs[(long long)s * HC + c] * alpha);
}

__global__ void add_bias(float* __restrict__ out, const float* __restrict__ bias, long long total) {
    long long i = (long long)blockIdx.x * blockDim.x + threadIdx.x;
    if (i < total) out[i] += bias[i & 127];
}

// ---------------- combine branches + per-graph mean pool ----------------
__global__ void combine_pool(const float* __restrict__ x1, const float* __restrict__ x2,
                             const int* __restrict__ batch, float* __restrict__ gsum,
                             float* __restrict__ gcnt, int N) {
    long long i = (long long)blockIdx.x * blockDim.x + threadIdx.x;
    if (i >= (long long)N * HC) return;
    int n = (int)(i >> 7), c = (int)(i & 127);
    float hv = (x1[i] + x2[i]) * 0.5f;
    int g = batch[n];
    atomicAdd(&gsum[g * HC + c], hv);
    if (c == 0) atomicAdd(&gcnt[g], 1.f);
}

// ---------------- FF head: relu(g@ffW+ffb) @ outW + outb -> sigmoid ----------------
__global__ void mlp_head(const float* __restrict__ gsum, const float* __restrict__ gcnt,
                         const float* __restrict__ ffW, const float* __restrict__ ffb,
                         const float* __restrict__ outW, const float* __restrict__ outb,
                         float* __restrict__ out) {
    int g = blockIdx.x, t = threadIdx.x;   // 128 threads
    __shared__ float gv[HC];
    __shared__ float red[HC];
    float cnt = fmaxf(gcnt[g], 1.f);
    gv[t] = gsum[g * HC + t] / cnt;
    __syncthreads();
    float acc = ffb[t];
    for (int k = 0; k < HC; ++k) acc += gv[k] * ffW[k * HC + t];
    acc = fmaxf(acc, 0.f);
    red[t] = acc * outW[t];
    __syncthreads();
    for (int s = 64; s > 0; s >>= 1) { if (t < s) red[t] += red[t + s]; __syncthreads(); }
    if (t == 0) out[g] = 1.f / (1.f + __expf(-(red[0] + outb[0])));
}

// =====================================================================
static inline unsigned blks(long long total, int tpb) { return (unsigned)((total + tpb - 1) / tpb); }

struct GatState {
    float *xs, *a_src, *a_dst, *denom, *lsum, *lcnt, *lattr, *ecoef;
    unsigned *amax;
    _Float16 *wfrag;
};

static void run_gat(const float* xin, int K, int reluA,
                    const float* Wm, const float* as_, const float* ad_, const float* bias,
                    const int* srcv, const int* dstv, long long E,
                    const float* attr, const float* We, const float* ae,
                    const GatState& st, float* outb, hipStream_t stream) {
    const int N = NN;
    // 1) xs = (relu?)xin @ W      (WMMA, pre-packed B fragments)
    {
        pack_wfrag<<<blks((long long)K * HC, 256), 256, 0, stream>>>(Wm, st.wfrag, K);
        long long thr = (long long)(N / 16) * 8 * 32;
        if (K == 32) {
            if (reluA) gemm_wmma<32, 1><<<blks(thr, 256), 256, 0, stream>>>(xin, st.wfrag, st.xs, N);
            else       gemm_wmma<32, 0><<<blks(thr, 256), 256, 0, stream>>>(xin, st.wfrag, st.xs, N);
        } else {
            if (reluA) gemm_wmma<128, 1><<<blks(thr, 256), 256, 0, stream>>>(xin, st.wfrag, st.xs, N);
            else       gemm_wmma<128, 0><<<blks(thr, 256), 256, 0, stream>>>(xin, st.wfrag, st.xs, N);
        }
    }
    // 2) attention coefficients
    attn_coeff<<<blks((long long)N * NH, 256), 256, 0, stream>>>(st.xs, as_, ad_, st.a_src, st.a_dst, N);
    // 3) edge-attr self-loop mean + collapsed edge coefficient
    if (attr) {
        hipMemsetAsync(st.lsum, 0, N * sizeof(float), stream);
        hipMemsetAsync(st.lcnt, 0, N * sizeof(float), stream);
        edge_attr_stats<<<blks(E, 256), 256, 0, stream>>>(dstv, attr, st.lsum, st.lcnt, (int)E);
        loop_attr_final<<<blks(N, 256), 256, 0, stream>>>(st.lsum, st.lcnt, st.lattr, N);
        compute_ecoef<<<1, NH, 0, stream>>>(We, ae, st.ecoef);
    }
    // 4) segment softmax state + output accumulator
    hipMemsetAsync(st.amax, 0, (size_t)N * NH * sizeof(unsigned), stream);
    hipMemsetAsync(st.denom, 0, (size_t)N * NH * sizeof(float), stream);
    hipMemsetAsync(outb, 0, (size_t)N * HC * sizeof(float), stream);
    long long nah = (E + N) * NH;
    alpha_max<<<blks(nah, 256), 256, 0, stream>>>(srcv, dstv, attr, st.lattr, st.a_src, st.a_dst,
                                                  st.ecoef, st.amax, E, N);
    alpha_denom<<<blks(nah, 256), 256, 0, stream>>>(srcv, dstv, attr, st.lattr, st.a_src, st.a_dst,
                                                    st.ecoef, st.amax, st.denom, E, N);
    // 5) scatter messages
    long long nmc = (E + N) * HC;
    message_pass<<<blks(nmc, 256), 256, 0, stream>>>(srcv, dstv, attr, st.lattr, st.a_src, st.a_dst,
                                                     st.ecoef, st.amax, st.denom, st.xs, outb, E, N);
    // 6) bias
    add_bias<<<blks((long long)N * HC, 256), 256, 0, stream>>>(outb, bias, (long long)N * HC);
}

extern "C" void kernel_launch(void* const* d_in, const int* in_sizes, int n_in,
                              void* d_out, int out_size, void* d_ws, size_t ws_size,
                              hipStream_t stream) {
    const float* x     = (const float*)d_in[0];
    const int*   cov   = (const int*)d_in[1];
    const int*   dist  = (const int*)d_in[2];
    const float* dattr = (const float*)d_in[3];
    const int*   batch = (const int*)d_in[4];
    const float *W_c1 = (const float*)d_in[5],  *as_c1 = (const float*)d_in[6],
                *ad_c1 = (const float*)d_in[7], *b_c1 = (const float*)d_in[8];
    const float *W_c2 = (const float*)d_in[9],  *as_c2 = (const float*)d_in[10],
                *ad_c2 = (const float*)d_in[11],*b_c2 = (const float*)d_in[12];
    const float *W_d1 = (const float*)d_in[13], *as_d1 = (const float*)d_in[14],
                *ad_d1 = (const float*)d_in[15],*b_d1 = (const float*)d_in[16];
    const float *W_d2 = (const float*)d_in[17], *as_d2 = (const float*)d_in[18],
                *ad_d2 = (const float*)d_in[19],*b_d2 = (const float*)d_in[20];
    const float *We_d1 = (const float*)d_in[21], *ae_d1 = (const float*)d_in[22];
    const float *We_d2 = (const float*)d_in[23], *ae_d2 = (const float*)d_in[24];
    const float *ffW = (const float*)d_in[25], *ffb = (const float*)d_in[26];
    const float *outW = (const float*)d_in[27], *outb = (const float*)d_in[28];
    float* out = (float*)d_out;

    long long Ec = in_sizes[1] / 2;
    long long Ed = in_sizes[2] / 2;
    const int* cov_src  = cov;            const int* cov_dst  = cov + Ec;
    const int* dist_src = dist;           const int* dist_dst = dist + Ed;

    const int N = NN;
    // ---- workspace carve-up (all offsets 32B-aligned) ----
    char* p = (char*)d_ws;
    float* ws_xs   = (float*)p; p += (size_t)N * HC * sizeof(float);
    float* bufP    = (float*)p; p += (size_t)N * HC * sizeof(float);
    float* bufQ    = (float*)p; p += (size_t)N * HC * sizeof(float);
    float* bufR    = (float*)p; p += (size_t)N * HC * sizeof(float);
    float* a_src   = (float*)p; p += (size_t)N * NH * sizeof(float);
    float* a_dst   = (float*)p; p += (size_t)N * NH * sizeof(float);
    unsigned* amax = (unsigned*)p; p += (size_t)N * NH * sizeof(unsigned);
    float* denom   = (float*)p; p += (size_t)N * NH * sizeof(float);
    float* lsum    = (float*)p; p += (size_t)N * sizeof(float);
    float* lcnt    = (float*)p; p += (size_t)N * sizeof(float);
    float* lattr   = (float*)p; p += (size_t)N * sizeof(float);
    float* ecoef   = (float*)p; p += 32 * sizeof(float);      // padded for alignment
    float* gsum    = (float*)p; p += (size_t)NG * HC * sizeof(float);
    float* gcnt    = (float*)p; p += (size_t)NG * sizeof(float);
    _Float16* wfrag = (_Float16*)p; p += (size_t)HC * HC * sizeof(_Float16);
    (void)ws_size; (void)n_in; (void)out_size;

    GatState st{ws_xs, a_src, a_dst, denom, lsum, lcnt, lattr, ecoef, amax, wfrag};

    // covalent branch: x1 = gat2(relu(gat1(x)))
    run_gat(x,    32,  0, W_c1, as_c1, ad_c1, b_c1, cov_src, cov_dst, Ec,
            nullptr, nullptr, nullptr, st, bufP, stream);           // bufP = gat1 (pre-relu)
    run_gat(bufP, HC,  1, W_c2, as_c2, ad_c2, b_c2, cov_src, cov_dst, Ec,
            nullptr, nullptr, nullptr, st, bufQ, stream);           // bufQ = x1

    // distance branch: x2 = gat2(relu(gat1(x)))  with edge attrs
    run_gat(x,    32,  0, W_d1, as_d1, ad_d1, b_d1, dist_src, dist_dst, Ed,
            dattr, We_d1, ae_d1, st, bufP, stream);                 // bufP = gat1 (pre-relu)
    run_gat(bufP, HC,  1, W_d2, as_d2, ad_d2, b_d2, dist_src, dist_dst, Ed,
            dattr, We_d2, ae_d2, st, bufR, stream);                 // bufR = x2

    // pool + head
    hipMemsetAsync(gsum, 0, (size_t)NG * HC * sizeof(float), stream);
    hipMemsetAsync(gcnt, 0, (size_t)NG * sizeof(float), stream);
    combine_pool<<<blks((long long)N * HC, 256), 256, 0, stream>>>(bufQ, bufR, batch, gsum, gcnt, N);
    mlp_head<<<NG, HC, 0, stream>>>(gsum, gcnt, ffW, ffb, outW, outb, out);
}
